// Attention_17841294148254
// MI455X (gfx1250) — matrix operand order repaired and checked by reading, not compiled, not run on hardware
//
#include <hip/hip_runtime.h>
#include <stdint.h>

// ---------------- types ----------------
typedef __attribute__((ext_vector_type(16))) __bf16 v16bf;
typedef __attribute__((ext_vector_type(8)))  __bf16 v8bf;
typedef __attribute__((ext_vector_type(4)))  __bf16 v4bf;
typedef __attribute__((ext_vector_type(8)))  float  v8f;
typedef __attribute__((ext_vector_type(4)))  uint32_t u32x4;
typedef __attribute__((ext_vector_type(8)))  uint32_t u32x8;

union bfrag16 { v16bf v; v8bf h[2]; };

static __device__ __forceinline__ __bf16 f2bf(float f) {
  union { float f; uint32_t u; } a{f};
  uint32_t r = a.u + 0x7FFFu + ((a.u >> 16) & 1u);
  union { uint16_t u; __bf16 b; } c{(uint16_t)(r >> 16)};
  return c.b;
}

// ---------------- Tensor Data Mover: 2D tile global->LDS ----------------
// Builds a D# (group0: 4 SGPRs, group1: 8 SGPRs) per CDNA5 ISA
// and issues tensor_load_to_lds. pad_* produce a padded LDS row stride.
static __device__ __forceinline__ void tdm_load_2d(
    uint32_t lds_off, const void* gptr,
    uint32_t tdim0, uint32_t tdim1, uint64_t stride0,
    uint32_t tile0, uint32_t tile1,
    uint32_t pad_int_code, uint32_t pad_amt_code, uint32_t pad_en) {
  uint64_t ga = (uint64_t)(uintptr_t)gptr;
  u32x4 g0;
  g0[0] = 1u;                                   // count=1 (valid user descriptor)
  g0[1] = lds_off;                              // LDS byte address
  g0[2] = (uint32_t)ga;                         // global_addr[31:0]  (bits 95:64)
  g0[3] = (uint32_t)((ga >> 32) & 0x01FFFFFFu)  // global_addr[56:32]
        | (2u << 30);                           // type=2 ("image")
  u32x8 g1;
  g1[0] = (1u << 16)                            // data_size=1 (2 bytes)
        | (pad_en << 20) | (pad_int_code << 22) | (pad_amt_code << 25);
  g1[1] = (tdim0 & 0xFFFFu) << 16;              // tensor_dim0[15:0] at bits 63:48
  g1[2] = (tdim0 >> 16) | ((tdim1 & 0xFFFFu) << 16);
  g1[3] = (tdim1 >> 16) | (tile0 << 16);        // tile_dim0 at bits 127:112
  g1[4] = tile1;                                // tile_dim1 (tile_dim2=0 -> 2D)
  g1[5] = (uint32_t)stride0;                    // tensor_dim0_stride[31:0]
  g1[6] = (uint32_t)(stride0 >> 32) & 0xFFFFu;  // stride0[47:32]; dim1_stride=0
  g1[7] = 0u;
  asm volatile("tensor_load_to_lds %0, %1" :: "s"(g0), "s"(g1) : "memory");
}

// ---------------- fp32 -> bf16 elementwise ----------------
__global__ __launch_bounds__(256) void cvt_bf16(const float* __restrict__ in,
                                                __bf16* __restrict__ out, size_t n) {
  size_t i = ((size_t)blockIdx.x * 256 + threadIdx.x) * 4;
  if (i >= n) return;
  float4 f = *(const float4*)(in + i);
  v4bf t = {f2bf(f.x), f2bf(f.y), f2bf(f.z), f2bf(f.w)};
  *(v4bf*)(out + i) = t;
}

// ---------------- fp32 [K][N] -> bf16 [N][K] transpose ----------------
__global__ __launch_bounds__(256) void cvt_t_bf16(const float* __restrict__ W,
                                                  __bf16* __restrict__ Wt,
                                                  int Kd, int Nd) {
  __shared__ float tile[32][33];
  const int bk = blockIdx.y * 32, bn = blockIdx.x * 32;
  const int tid = threadIdx.x;
  for (int i = 0; i < 4; ++i) {
    int idx = tid + i * 256; int r = idx >> 5, c = idx & 31;
    tile[r][c] = W[(size_t)(bk + r) * Nd + bn + c];
  }
  __syncthreads();
  for (int i = 0; i < 4; ++i) {
    int idx = tid + i * 256; int r = idx >> 5, c = idx & 31;
    Wt[(size_t)(bn + r) * Kd + bk + c] = f2bf(tile[c][r]);
  }
}

// ---------------- GEMM: C[MxN] fp32 = A[MxK]bf16 @ (Bt[NxK]bf16)^T ---------
// 128x128 block tile, 8 waves (2x4), wave 64x32 (4x2 wmma accumulators).
// Tiles staged by TDM (double-buffered, padded to stride-40 rows in LDS).
__global__ __launch_bounds__(256) void gemm_bf16(
    const __bf16* __restrict__ A, const __bf16* __restrict__ Bt,
    float* __restrict__ C, int M, int N, int K) {
  __shared__ __bf16 As[2][128 * 40];
  __shared__ __bf16 Bs[2][128 * 40];

  const int tid  = threadIdx.x;
  const int lane = tid & 31;
  const int wave = tid >> 5;
  const int wm   = wave >> 2;
  const int wn   = wave & 3;
  const int half = lane >> 4;
  const int l16  = lane & 15;
  const int bm   = blockIdx.y * 128;
  const int bn   = blockIdx.x * 128;
  // uniform branch: exactly one wave drives the TDM (TDM ignores EXEC)
  const bool tdm = __builtin_amdgcn_readfirstlane(tid) < 32;

  v8f acc[4][2] = {};

  if (tdm) {  // prime tile 0: rows of 32 bf16 = 16 dwords, pad 4 dwords -> stride 40
    tdm_load_2d((uint32_t)(uintptr_t)&As[0][0], A + (size_t)bm * K,
                (uint32_t)K, (uint32_t)M, (uint64_t)K, 32, 128, 3, 3, 1);
    tdm_load_2d((uint32_t)(uintptr_t)&Bs[0][0], Bt + (size_t)bn * K,
                (uint32_t)K, (uint32_t)N, (uint64_t)K, 32, 128, 3, 3, 1);
  }

  int p = 0;
  for (int k0 = 0; k0 < K; k0 += 32, p ^= 1) {
    if (tdm) {
      if (k0 + 32 < K) {  // overlap next tile's DMA with this tile's WMMA
        tdm_load_2d((uint32_t)(uintptr_t)&As[p ^ 1][0],
                    A + (size_t)bm * K + (k0 + 32),
                    (uint32_t)K, (uint32_t)M, (uint64_t)K, 32, 128, 3, 3, 1);
        tdm_load_2d((uint32_t)(uintptr_t)&Bs[p ^ 1][0],
                    Bt + (size_t)bn * K + (k0 + 32),
                    (uint32_t)K, (uint32_t)N, (uint64_t)K, 32, 128, 3, 3, 1);
        __builtin_amdgcn_s_wait_tensorcnt(2);   // wait for tile k only
      } else {
        __builtin_amdgcn_s_wait_tensorcnt(0);
      }
    }
    __syncthreads();

    const __bf16* as = As[p];
    const __bf16* bs = Bs[p];
    bfrag16 bf[2];
    for (int j = 0; j < 2; ++j) {
      int col = wn * 32 + j * 16 + l16;
      bf[j].h[0] = *(const v8bf*)&bs[col * 40 + half * 8];
      bf[j].h[1] = *(const v8bf*)&bs[col * 40 + 16 + half * 8];
    }
    for (int i = 0; i < 4; ++i) {
      int row = wm * 64 + i * 16 + l16;
      bfrag16 af;
      af.h[0] = *(const v8bf*)&as[row * 40 + half * 8];
      af.h[1] = *(const v8bf*)&as[row * 40 + 16 + half * 8];
      for (int j = 0; j < 2; ++j) {
        acc[i][j] = __builtin_amdgcn_wmma_f32_16x16x32_bf16(
            false, af.v, false, bf[j].v, (short)0, acc[i][j], false, false);
      }
    }
    __syncthreads();
  }

  for (int i = 0; i < 4; ++i)
    for (int j = 0; j < 2; ++j) {
      int col = bn + wn * 32 + j * 16 + l16;
      for (int r = 0; r < 8; ++r) {
        int row = bm + wm * 64 + i * 16 + r + 8 * half;
        C[(size_t)row * N + col] = acc[i][j][r];
      }
    }
}

// ---------------- RoPE + relayout to head-major bf16 -----------------------
__global__ __launch_bounds__(256) void rope_q(
    const float* __restrict__ q, const float* __restrict__ cosr,
    const float* __restrict__ sinr, __bf16* __restrict__ out, int S) {
  size_t i = (size_t)blockIdx.x * blockDim.x + threadIdx.x;
  int d2 = (int)(i & 63); size_t t = i >> 6;
  int h = (int)(t & 31); int s = (int)(t >> 5);
  if (s >= S) return;
  const float* src = q + (size_t)s * 4096 + h * 128 + 2 * d2;
  float x1 = src[0], x2 = src[1];
  float c = cosr[s * 64 + d2], sn = sinr[s * 64 + d2];
  const float sc = 0.08838834764831845f;  // 1/sqrt(128) folded into Q
  __bf16* o = out + ((size_t)h * S + s) * 128 + 2 * d2;
  o[0] = f2bf((x1 * c - x2 * sn) * sc);
  o[1] = f2bf((x1 * sn + x2 * c) * sc);
}

__global__ __launch_bounds__(256) void rope_k(
    const float* __restrict__ k, const float* __restrict__ cosr,
    const float* __restrict__ sinr, __bf16* __restrict__ out, int S) {
  size_t i = (size_t)blockIdx.x * blockDim.x + threadIdx.x;
  int d2 = (int)(i & 63); size_t t = i >> 6;
  int h = (int)(t & 7); int s = (int)(t >> 3);
  if (s >= S) return;
  const float* src = k + (size_t)s * 1024 + h * 128 + 2 * d2;
  float x1 = src[0], x2 = src[1];
  float c = cosr[s * 64 + d2], sn = sinr[s * 64 + d2];
  __bf16* o = out + ((size_t)h * S + s) * 128 + 2 * d2;
  o[0] = f2bf(x1 * c - x2 * sn);
  o[1] = f2bf(x1 * sn + x2 * c);
}

__global__ __launch_bounds__(256) void conv_v(
    const float* __restrict__ v, __bf16* __restrict__ out, int S) {
  size_t i = (size_t)blockIdx.x * blockDim.x + threadIdx.x;
  int d = (int)(i & 127); size_t t = i >> 7;
  int h = (int)(t & 7); int s = (int)(t >> 3);
  if (s >= S) return;
  out[((size_t)h * S + s) * 128 + d] = f2bf(v[(size_t)s * 1024 + h * 128 + d]);
}

// ---------------- fused causal flash attention (GQA 4:1) --------------------
__global__ __launch_bounds__(256) void attn_fa(
    const __bf16* __restrict__ Qh,   // [32][S][128]
    const __bf16* __restrict__ Kh,   // [8][S][128]
    const __bf16* __restrict__ Vh,   // [8][S][128]
    __bf16* __restrict__ O,          // [S][4096] bf16 (feeds out-proj GEMM)
    int S) {
  __shared__ __bf16 Klds[32 * 136];      // [key][d], stride 136 (TDM pad)
  __shared__ __bf16 Vt[128 * 40];        // [d][key], stride 40
  __shared__ __bf16 Plds[8 * 16 * 40];   // per-wave P scratch

  const int tid  = threadIdx.x;
  const int lane = tid & 31;
  const int wave = tid >> 5;
  const int half = lane >> 4;
  const int l16  = lane & 15;
  const int head = blockIdx.y;
  const int kv   = head >> 2;
  const int qbase = blockIdx.x * 128 + wave * 16;
  const bool tdm = __builtin_amdgcn_readfirstlane(tid) < 32;

  bfrag16 qf[4];
  const __bf16* qrow = Qh + ((size_t)head * S + qbase + l16) * 128;
  for (int d = 0; d < 4; ++d) {
    qf[d].h[0] = *(const v8bf*)(qrow + d * 32 + half * 8);
    qf[d].h[1] = *(const v8bf*)(qrow + d * 32 + 16 + half * 8);
  }

  v8f o_acc[8] = {};
  float m_i[8], l_i[8];
  for (int r = 0; r < 8; ++r) { m_i[r] = -1e30f; l_i[r] = 0.0f; }

  const int kb_end = blockIdx.x * 128 + 128;
  for (int kb = 0; kb < kb_end; kb += 32) {
    // K tile via TDM: rows of 128 bf16 = 64 dwords (code 5), pad 4 dwords -> 136
    if (tdm)
      tdm_load_2d((uint32_t)(uintptr_t)Klds, Kh + ((size_t)kv * S + kb) * 128,
                  128u, (uint32_t)S, 128ull, 128, 32, 5, 3, 1);
    // V tile staged transposed by all threads
    for (int i = 0; i < 2; ++i) {
      int idx = tid + i * 256;
      int key = idx >> 4;
      int dd  = (idx & 15) * 8;
      v8bf vv = *(const v8bf*)(Vh + ((size_t)kv * S + kb + key) * 128 + dd);
      for (int e = 0; e < 8; ++e) Vt[(dd + e) * 40 + key] = vv[e];
    }
    if (tdm) __builtin_amdgcn_s_wait_tensorcnt(0);
    __syncthreads();

    v8f s0 = {}, s1 = {};
    for (int d = 0; d < 4; ++d) {
      bfrag16 k0f, k1f;
      k0f.h[0] = *(const v8bf*)&Klds[l16 * 136 + d * 32 + half * 8];
      k0f.h[1] = *(const v8bf*)&Klds[l16 * 136 + d * 32 + 16 + half * 8];
      k1f.h[0] = *(const v8bf*)&Klds[(16 + l16) * 136 + d * 32 + half * 8];
      k1f.h[1] = *(const v8bf*)&Klds[(16 + l16) * 136 + d * 32 + 16 + half * 8];
      s0 = __builtin_amdgcn_wmma_f32_16x16x32_bf16(false, qf[d].v, false, k0f.v,
                                                   (short)0, s0, false, false);
      s1 = __builtin_amdgcn_wmma_f32_16x16x32_bf16(false, qf[d].v, false, k1f.v,
                                                   (short)0, s1, false, false);
    }

    const int key0 = kb + l16;
    const int key1 = kb + 16 + l16;
    float p0[8], p1[8];
    for (int r = 0; r < 8; ++r) {
      int qr = qbase + r + 8 * half;
      float v0 = (key0 <= qr) ? s0[r] : -1e30f;
      float v1 = (key1 <= qr) ? s1[r] : -1e30f;
      float mrow = fmaxf(v0, v1);
      for (int off = 1; off < 16; off <<= 1)
        mrow = fmaxf(mrow, __shfl_xor(mrow, off, 16));
      float mnew = fmaxf(m_i[r], mrow);
      float corr = __expf(m_i[r] - mnew);
      p0[r] = __expf(v0 - mnew);
      p1[r] = __expf(v1 - mnew);
      float rowsum = p0[r] + p1[r];
      for (int off = 1; off < 16; off <<= 1)
        rowsum += __shfl_xor(rowsum, off, 16);
      l_i[r] = l_i[r] * corr + rowsum;
      m_i[r] = mnew;
      for (int n = 0; n < 8; ++n) o_acc[n][r] *= corr;
    }

    __bf16* pw = &Plds[wave * 16 * 40];
    for (int r = 0; r < 8; ++r) {
      pw[(r + 8 * half) * 40 + l16]      = f2bf(p0[r]);
      pw[(r + 8 * half) * 40 + 16 + l16] = f2bf(p1[r]);
    }
    bfrag16 pf;
    pf.h[0] = *(const v8bf*)&pw[l16 * 40 + half * 8];
    pf.h[1] = *(const v8bf*)&pw[l16 * 40 + 16 + half * 8];

    for (int n = 0; n < 8; ++n) {
      bfrag16 vf;
      int col = n * 16 + l16;
      vf.h[0] = *(const v8bf*)&Vt[col * 40 + half * 8];
      vf.h[1] = *(const v8bf*)&Vt[col * 40 + 16 + half * 8];
      o_acc[n] = __builtin_amdgcn_wmma_f32_16x16x32_bf16(false, pf.v, false, vf.v,
                                                         (short)0, o_acc[n], false, false);
    }
    __syncthreads();
  }

  for (int r = 0; r < 8; ++r) {
    int row = qbase + r + 8 * half;
    float inv = 1.0f / l_i[r];
    for (int n = 0; n < 8; ++n)
      O[(size_t)row * 4096 + head * 128 + n * 16 + l16] = f2bf(o_acc[n][r] * inv);
  }
}

// ---------------- host launch ----------------
extern "C" void kernel_launch(void* const* d_in, const int* in_sizes, int n_in,
                              void* d_out, int out_size, void* d_ws, size_t ws_size,
                              hipStream_t stream) {
  const int S = 2048;
  const float* hid  = (const float*)d_in[0];
  const float* cosr = (const float*)d_in[1];
  const float* sinr = (const float*)d_in[2];
  const float* wq = (const float*)d_in[5];
  const float* wk = (const float*)d_in[6];
  const float* wv = (const float*)d_in[7];
  const float* wo = (const float*)d_in[8];
  float* out = (float*)d_out;

  // workspace carve-up
  float* qf32 = (float*)d_ws;                        // S*4096 f32
  float* kf32 = qf32 + (size_t)S * 4096;             // S*1024 f32
  float* vf32 = kf32 + (size_t)S * 1024;             // S*1024 f32
  __bf16* hb  = (__bf16*)(vf32 + (size_t)S * 1024);  // S*4096
  __bf16* qb  = hb  + (size_t)S * 4096;              // 32*S*128
  __bf16* kbh = qb  + (size_t)S * 4096;              // 8*S*128
  __bf16* vbh = kbh + (size_t)S * 1024;              // 8*S*128
  __bf16* aob = vbh + (size_t)S * 1024;              // S*4096
  __bf16* wqt = aob + (size_t)S * 4096;              // 4096*4096
  __bf16* wkt = wqt + (size_t)4096 * 4096;           // 1024*4096
  __bf16* wvt = wkt + (size_t)1024 * 4096;           // 1024*4096
  __bf16* wot = wvt + (size_t)1024 * 4096;           // 4096*4096

  dim3 blk(256);
  // one-time bf16 conversion / weight transposition
  cvt_bf16<<<8192, blk, 0, stream>>>(hid, hb, (size_t)S * 4096);
  cvt_t_bf16<<<dim3(128, 128), blk, 0, stream>>>(wq, wqt, 4096, 4096);
  cvt_t_bf16<<<dim3(32, 128),  blk, 0, stream>>>(wk, wkt, 4096, 1024);
  cvt_t_bf16<<<dim3(32, 128),  blk, 0, stream>>>(wv, wvt, 4096, 1024);
  cvt_t_bf16<<<dim3(128, 128), blk, 0, stream>>>(wo, wot, 4096, 4096);
  // QKV projections (bf16 WMMA, fp32 accum)
  gemm_bf16<<<dim3(32, 16), blk, 0, stream>>>(hb, wqt, qf32, S, 4096, 4096);
  gemm_bf16<<<dim3(8, 16),  blk, 0, stream>>>(hb, wkt, kf32, S, 1024, 4096);
  gemm_bf16<<<dim3(8, 16),  blk, 0, stream>>>(hb, wvt, vf32, S, 1024, 4096);
  // RoPE + head-major bf16 relayout
  rope_q<<<16384, blk, 0, stream>>>(qf32, cosr, sinr, qb, S);
  rope_k<<<4096, blk, 0, stream>>>(kf32, cosr, sinr, kbh, S);
  conv_v<<<8192, blk, 0, stream>>>(vf32, vbh, S);
  // fused causal attention
  attn_fa<<<dim3(16, 32), blk, 0, stream>>>(qb, kbh, vbh, aob, S);
  // output projection
  gemm_bf16<<<dim3(32, 16), blk, 0, stream>>>(aob, wot, out, S, 4096, 4096);
}
